// BertNer_ORG_10359461118296
// MI455X (gfx1250) — matrix operand order repaired and checked by reading, not compiled, not run on hardware
//
#include <hip/hip_runtime.h>
#include <hip/hip_bf16.h>
#include <math.h>

// ---------------- problem constants ----------------
#define B_  8
#define S_  256
#define H_  768
#define L_  12
#define NH_ 12
#define FF_ 3072
#define NL_ 9
#define D_  64
#define T_  (B_*S_)     // 2048 tokens

typedef __attribute__((ext_vector_type(16))) _Float16 v16h;
typedef __attribute__((ext_vector_type(8)))  _Float16 v8h;
typedef __attribute__((ext_vector_type(4)))  _Float16 v4h;
typedef __attribute__((ext_vector_type(2)))  _Float16 v2h;
typedef __attribute__((ext_vector_type(8)))  float    v8f;

union FragH { v16h v; v8h half2[2]; _Float16 h[16]; };

// ---------------- helpers ----------------
__device__ __forceinline__ float blk_sum(float v, float* red) {
    int t = threadIdx.x;
    red[t] = v; __syncthreads();
    for (int o = 128; o > 0; o >>= 1) { if (t < o) red[t] += red[t + o]; __syncthreads(); }
    float r = red[0]; __syncthreads();
    return r;
}
__device__ __forceinline__ float blk_max(float v, float* red) {
    int t = threadIdx.x;
    red[t] = v; __syncthreads();
    for (int o = 128; o > 0; o >>= 1) { if (t < o) red[t] = fmaxf(red[t], red[t + o]); __syncthreads(); }
    float r = red[0]; __syncthreads();
    return r;
}
__device__ __forceinline__ float gelu_exact(float x) {
    return 0.5f * x * (1.0f + erff(x * 0.70710678118654752f));
}

// ---------------- f32 -> f16 conversion ----------------
__global__ void k_cvt_f16(const float* __restrict__ src, _Float16* __restrict__ dst, long long n) {
    long long i = (long long)blockIdx.x * blockDim.x + threadIdx.x;
    long long stride = (long long)gridDim.x * blockDim.x;
    for (; i < n; i += stride) dst[i] = (_Float16)src[i];
}

// ---------------- attention mask bias ----------------
__global__ void k_maskbias(const int* __restrict__ mask, float* __restrict__ mb, int n) {
    int i = blockIdx.x * blockDim.x + threadIdx.x;
    if (i < n) mb[i] = (1.0f - (float)mask[i]) * -1.0e4f;
}

// ---------------- embeddings + LayerNorm ----------------
__global__ __launch_bounds__(256) void k_embed(
    const int* __restrict__ ids, const int* __restrict__ tids,
    const float* __restrict__ wemb, const float* __restrict__ pemb,
    const float* __restrict__ temb, const float* __restrict__ g,
    const float* __restrict__ bta, float* __restrict__ out, _Float16* __restrict__ out16) {
    __shared__ float red[256];
    int tok = blockIdx.x;
    int s = tok % S_;
    int wid = ids[tok], tyid = tids[tok];
    float v[3]; float sum = 0.f;
    #pragma unroll
    for (int i = 0; i < 3; ++i) {
        int idx = i * 256 + threadIdx.x;
        float val = wemb[(long long)wid * H_ + idx] + pemb[(long long)s * H_ + idx]
                  + temb[(long long)tyid * H_ + idx];
        v[i] = val; sum += val;
    }
    float mean = blk_sum(sum, red) * (1.0f / H_);
    float sq = 0.f;
    #pragma unroll
    for (int i = 0; i < 3; ++i) { float d = v[i] - mean; sq += d * d; }
    float var = blk_sum(sq, red) * (1.0f / H_);
    float rstd = rsqrtf(var + 1e-12f);
    #pragma unroll
    for (int i = 0; i < 3; ++i) {
        int idx = i * 256 + threadIdx.x;
        float o = (v[i] - mean) * rstd * g[idx] + bta[idx];
        out[(long long)tok * H_ + idx] = o;
        out16[(long long)tok * H_ + idx] = (_Float16)o;
    }
}

// ---------------- residual + LayerNorm ----------------
__global__ __launch_bounds__(256) void k_ln(
    const float* __restrict__ y, const float* __restrict__ res,
    const float* __restrict__ g, const float* __restrict__ bta,
    float* __restrict__ out, _Float16* __restrict__ out16) {
    __shared__ float red[256];
    long long base = (long long)blockIdx.x * H_;
    float v[3]; float sum = 0.f;
    #pragma unroll
    for (int i = 0; i < 3; ++i) {
        int idx = i * 256 + threadIdx.x;
        float val = y[base + idx] + res[base + idx];
        v[i] = val; sum += val;
    }
    float mean = blk_sum(sum, red) * (1.0f / H_);
    float sq = 0.f;
    #pragma unroll
    for (int i = 0; i < 3; ++i) { float d = v[i] - mean; sq += d * d; }
    float var = blk_sum(sq, red) * (1.0f / H_);
    float rstd = rsqrtf(var + 1e-12f);
    #pragma unroll
    for (int i = 0; i < 3; ++i) {
        int idx = i * 256 + threadIdx.x;
        float o = (v[i] - mean) * rstd * g[idx] + bta[idx];
        out[base + idx] = o;
        out16[base + idx] = (_Float16)o;
    }
}

// ---------------- generic batched WMMA GEMM ----------------
// C[z] = act( alpha * A[z] @ B[z] + bias ), optional f32 and/or f16 outputs.
// Block tile 128(M) x 64(N), 8 wave32; each wave owns a 32x32 quadrant (4 WMMAs/K-step).
// Double-buffered LDS staging: global loads for step s+1 issue before step s's WMMAs.
struct GemmP {
    const _Float16* A;  long long lda, sAz;
    const _Float16* Bm; long long ldb, sBz;
    float*     C;    long long ldc;   long long sCb,   sCh;
    _Float16*  C16;  long long ldc16; long long sC16b, sC16h;
    int divC;                    // f32/f16 base = (z/divC)*sC*b + (z%divC)*sC*h
    int c16mode;                 // 0 normal, 1 Q/V head split [B*NH,S,D], 2 K^T head split [B*NH,D,S]
    const float* bias; int bias_mode; // 0 none, 1 per-N vector, 2 attn-mask [B,S] via z/NH_
    int M, N, K;
    float alpha;
    int act;                     // 0 none, 1 gelu(exact)
};

#define LDS_STRIDE 40   // f16 elements per LDS row (32 data + 8 pad) -> 80B, 16B aligned

template <bool EXACT>   // EXACT: M%128==0, N%64==0, lda/ldb%8==0 -> no guards
__global__ __launch_bounds__(256) void k_gemm(GemmP p) {
    __shared__ _Float16 As[2][128 * LDS_STRIDE];  // row-major:  As[m][k]
    __shared__ _Float16 Bt[2][64 * LDS_STRIDE];   // transposed: Bt[n][k]
    const int tid  = threadIdx.x;
    const int lane = tid & 31;
    const int wave = tid >> 5;
    const int wmi = wave & 3;      // 4 waves along M (32 rows each)
    const int wni = wave >> 2;     // 2 waves along N (32 cols each)
    const int z  = blockIdx.z;
    const int m0 = blockIdx.y * 128;
    const int n0 = blockIdx.x * 64;
    const _Float16* Ab = p.A  + (long long)z * p.sAz;
    const _Float16* Bb = p.Bm + (long long)z * p.sBz;

    v8f acc[2][2] = {};
    const int g  = lane >> 4;
    const int lm = lane & 15;

    // staging coordinates
    const int ar  = tid >> 2;            // 0..63 : A row (also +64)
    const int ac  = (tid & 3) * 8;       // 0,8,16,24 : A k-chunk
    const int bn  = (tid & 15) * 4;      // 0..60 : B n-chunk (4 cols)
    const int bk2 = (tid >> 4) * 2;      // 0..30 : B k-pair

    // register staging for next tile
    v8h ra0, ra1; v4h rb0, rb1;

    auto load_tile = [&](int k0) {
        if (EXACT || (m0 + ar) < p.M)
            ra0 = *(const v8h*)(Ab + (long long)(m0 + ar) * p.lda + (k0 + ac));
        else ra0 = (v8h)(_Float16)0.0f;
        if (EXACT || (m0 + ar + 64) < p.M)
            ra1 = *(const v8h*)(Ab + (long long)(m0 + ar + 64) * p.lda + (k0 + ac));
        else ra1 = (v8h)(_Float16)0.0f;
        if (EXACT) {
            rb0 = *(const v4h*)(Bb + (long long)(k0 + bk2) * p.ldb + (n0 + bn));
            rb1 = *(const v4h*)(Bb + (long long)(k0 + bk2 + 1) * p.ldb + (n0 + bn));
        } else {
            #pragma unroll
            for (int j = 0; j < 4; ++j) {
                int gn = n0 + bn + j;
                rb0[j] = (gn < p.N) ? Bb[(long long)(k0 + bk2) * p.ldb + gn] : (_Float16)0.0f;
                rb1[j] = (gn < p.N) ? Bb[(long long)(k0 + bk2 + 1) * p.ldb + gn] : (_Float16)0.0f;
            }
        }
    };
    auto store_tile = [&](int buf) {
        *(v8h*)(As[buf] + ar * LDS_STRIDE + ac) = ra0;
        *(v8h*)(As[buf] + (ar + 64) * LDS_STRIDE + ac) = ra1;
        #pragma unroll
        for (int j = 0; j < 4; ++j) {
            v2h pr; pr[0] = rb0[j]; pr[1] = rb1[j];
            *(v2h*)(Bt[buf] + (bn + j) * LDS_STRIDE + bk2) = pr;   // 4B ds_store
        }
    };

    // prologue: stage tile 0
    load_tile(0);
    store_tile(0);
    __syncthreads();

    const int nsteps = p.K >> 5;   // K is always a multiple of 32
    for (int s = 0; s < nsteps; ++s) {
        const int cur = s & 1;
        const bool more = (s + 1) < nsteps;
        if (more) load_tile((s + 1) * 32);   // issue global loads early

        // ---- A fragments (two 16-row tiles of this wave's 32 rows) ----
        FragH a0, a1;
        {
            const _Float16* r0 = As[cur] + (32 * wmi + lm) * LDS_STRIDE;
            const _Float16* r1 = As[cur] + (32 * wmi + 16 + lm) * LDS_STRIDE;
            a0.half2[0] = *(const v8h*)(r0 + 8 * g);
            a0.half2[1] = *(const v8h*)(r0 + 16 + 8 * g);
            a1.half2[0] = *(const v8h*)(r1 + 8 * g);
            a1.half2[1] = *(const v8h*)(r1 + 16 + 8 * g);
        }
        // ---- B fragments (two 16-col tiles of this wave's 32 cols) ----
        FragH b0, b1;
        {
            const _Float16* c0p = Bt[cur] + (32 * wni + lm) * LDS_STRIDE + 8 * g;
            const _Float16* c1p = Bt[cur] + (32 * wni + 16 + lm) * LDS_STRIDE + 8 * g;
            b0.half2[0] = *(const v8h*)(c0p);
            b0.half2[1] = *(const v8h*)(c0p + 16);
            b1.half2[0] = *(const v8h*)(c1p);
            b1.half2[1] = *(const v8h*)(c1p + 16);
        }
        acc[0][0] = __builtin_amdgcn_wmma_f32_16x16x32_f16(false, a0.v, false, b0.v, (short)0, acc[0][0], false, false);
        acc[0][1] = __builtin_amdgcn_wmma_f32_16x16x32_f16(false, a0.v, false, b1.v, (short)0, acc[0][1], false, false);
        acc[1][0] = __builtin_amdgcn_wmma_f32_16x16x32_f16(false, a1.v, false, b0.v, (short)0, acc[1][0], false, false);
        acc[1][1] = __builtin_amdgcn_wmma_f32_16x16x32_f16(false, a1.v, false, b1.v, (short)0, acc[1][1], false, false);

        if (more) {
            store_tile(cur ^ 1);   // fill the other buffer
            __syncthreads();       // one barrier per K-step
        }
    }

    // ---- epilogue ----
    long long cbase   = (long long)(z / p.divC) * p.sCb   + (long long)(z % p.divC) * p.sCh;
    long long cbase16 = (long long)(z / p.divC) * p.sC16b + (long long)(z % p.divC) * p.sC16h;
    #pragma unroll
    for (int tm = 0; tm < 2; ++tm) {
        #pragma unroll
        for (int tn = 0; tn < 2; ++tn) {
            int col = n0 + 32 * wni + 16 * tn + lm;
            if (!EXACT && col >= p.N) continue;
            float bias = 0.f;
            if (p.bias_mode == 1)      bias = p.bias[col];
            else if (p.bias_mode == 2) bias = p.bias[(long long)(z / NH_) * p.N + col];
            #pragma unroll
            for (int r = 0; r < 8; ++r) {
                int row = m0 + 32 * wmi + 16 * tm + r + 8 * g;  // C layout: VGPR r -> M = r + 8*laneGroup
                if (!EXACT && row >= p.M) continue;
                float val = acc[tm][tn][r] * p.alpha + bias;
                if (p.act == 1) val = gelu_exact(val);
                if (p.C)   p.C[cbase + (long long)row * p.ldc + col] = val;
                if (p.C16) {
                    long long idx;
                    if (p.c16mode == 0) {
                        idx = cbase16 + (long long)row * p.ldc16 + col;
                    } else {
                        int bb = row / S_, ss = row % S_;
                        int hh = col / D_, dd = col % D_;
                        long long zh = (long long)bb * NH_ + hh;
                        idx = (p.c16mode == 1)
                            ? (zh * S_ + ss) * (long long)D_ + dd      // [B*NH, S, D]
                            : (zh * D_ + dd) * (long long)S_ + ss;     // [B*NH, D, S]
                    }
                    p.C16[idx] = (_Float16)val;
                }
            }
        }
    }
}

// ---------------- row softmax (n <= 256) ----------------
__global__ __launch_bounds__(256) void k_softmax(
    const float* __restrict__ in, long long ldin,
    _Float16* __restrict__ out16, float* __restrict__ out32, long long ldout, int n) {
    __shared__ float red[256];
    long long row = blockIdx.x;
    int t = threadIdx.x;
    float v = (t < n) ? in[row * ldin + t] : -3.4e38f;
    float mx = blk_max(v, red);
    float e = (t < n) ? __expf(v - mx) : 0.f;
    float s = blk_sum(e, red);
    if (t < n) {
        float r = e / s;
        if (out16) out16[row * ldout + t] = (_Float16)r;
        if (out32) out32[row * ldout + t] = r;
    }
}

// ---------------- valid-mask stream compaction ----------------
__global__ __launch_bounds__(256) void k_compact(
    const int* __restrict__ valid, const _Float16* __restrict__ x16,
    _Float16* __restrict__ comp16) {
    __shared__ int pre[256];
    int b = blockIdx.x, s = threadIdx.x;
    int v = valid[b * S_ + s];
    pre[s] = v; __syncthreads();
    for (int o = 1; o < 256; o <<= 1) {
        int t2 = (s >= o) ? pre[s - o] : 0;
        __syncthreads();
        pre[s] += t2;
        __syncthreads();
    }
    int excl = pre[s] - v;
    long long base = (long long)b * S_ * H_;
    // phase 1: zero all rows of this batch
    {
        v8h zz = (v8h)(_Float16)0.0f;
        v8h* dst = (v8h*)(comp16 + base + (long long)s * H_);
        for (int c = 0; c < H_ / 8; ++c) dst[c] = zz;
    }
    __syncthreads();
    // phase 2: scatter valid rows to the front (stable)
    if (v) {
        const v8h* src = (const v8h*)(x16 + base + (long long)s * H_);
        v8h* dst = (v8h*)(comp16 + base + (long long)excl * H_);
        for (int c = 0; c < H_ / 8; ++c) dst[c] = src[c];
    }
}

// ---------------- host orchestration ----------------
extern "C" void kernel_launch(void* const* d_in, const int* in_sizes, int n_in,
                              void* d_out, int out_size, void* d_ws, size_t ws_size,
                              hipStream_t stream) {
    const int*   ids   = (const int*)d_in[0];
    const int*   imask = (const int*)d_in[1];
    const int*   tids  = (const int*)d_in[2];
    const int*   vmask = (const int*)d_in[3];
    const float* wemb  = (const float*)d_in[4];
    const float* pemb  = (const float*)d_in[5];
    const float* temb  = (const float*)d_in[6];
    const float* eg    = (const float*)d_in[7];
    const float* ebt   = (const float*)d_in[8];
    const float* Wq    = (const float*)d_in[9];  const float* bq  = (const float*)d_in[10];
    const float* Wk    = (const float*)d_in[11]; const float* bk  = (const float*)d_in[12];
    const float* Wv    = (const float*)d_in[13]; const float* bv  = (const float*)d_in[14];
    const float* Wo    = (const float*)d_in[15]; const float* bo  = (const float*)d_in[16];
    const float* g1    = (const float*)d_in[17]; const float* b1  = (const float*)d_in[18];
    const float* Wi    = (const float*)d_in[19]; const float* bi  = (const float*)d_in[20];
    const float* Wo2   = (const float*)d_in[21]; const float* bo2 = (const float*)d_in[22];
    const float* g2    = (const float*)d_in[23]; const float* b2  = (const float*)d_in[24];
    const float* clfW  = (const float*)d_in[25]; const float* clfb = (const float*)d_in[26];

    // bump allocator over workspace
    char* wsp = (char*)d_ws;
    size_t off = 0;
    auto alloc = [&](size_t bytes) -> void* {
        void* p = (void*)(wsp + off);
        off = (off + bytes + 255) & ~(size_t)255;
        return p;
    };

    const long long HH  = (long long)H_ * H_;
    const long long HFF = (long long)H_ * FF_;

    _Float16* wq16  = (_Float16*)alloc((size_t)L_ * HH  * 2);
    _Float16* wk16  = (_Float16*)alloc((size_t)L_ * HH  * 2);
    _Float16* wv16  = (_Float16*)alloc((size_t)L_ * HH  * 2);
    _Float16* wo16  = (_Float16*)alloc((size_t)L_ * HH  * 2);
    _Float16* wi16  = (_Float16*)alloc((size_t)L_ * HFF * 2);
    _Float16* wo216 = (_Float16*)alloc((size_t)L_ * HFF * 2);
    _Float16* clfW16 = (_Float16*)alloc((size_t)H_ * NL_ * 2);

    float*    mb    = (float*)alloc((size_t)B_ * S_ * 4);
    float*    x     = (float*)alloc((size_t)T_ * H_ * 4);
    _Float16* x16   = (_Float16*)alloc((size_t)T_ * H_ * 2);
    float*    tmp   = (float*)alloc((size_t)T_ * H_ * 4);
    _Float16* q16   = (_Float16*)alloc((size_t)T_ * H_ * 2);
    _Float16* k16T  = (_Float16*)alloc((size_t)T_ * H_ * 2);
    _Float16* v16b  = (_Float16*)alloc((size_t)T_ * H_ * 2);
    float*    scores = (float*)alloc((size_t)B_ * NH_ * S_ * S_ * 4);
    _Float16* probs  = (_Float16*)alloc((size_t)B_ * NH_ * S_ * S_ * 2);
    float*    ctx    = (float*)alloc((size_t)T_ * H_ * 4);
    _Float16* ctx16  = (_Float16*)alloc((size_t)T_ * H_ * 2);
    _Float16* ff16   = (_Float16*)alloc((size_t)T_ * FF_ * 2);
    _Float16* comp16 = (_Float16*)alloc((size_t)T_ * H_ * 2);
    float*    logits = (float*)alloc((size_t)T_ * NL_ * 4);
    (void)ws_size; (void)n_in; (void)in_sizes; (void)out_size;

    // weight conversion to f16
    k_cvt_f16<<<2048, 256, 0, stream>>>(Wq,  wq16,  (long long)L_ * HH);
    k_cvt_f16<<<2048, 256, 0, stream>>>(Wk,  wk16,  (long long)L_ * HH);
    k_cvt_f16<<<2048, 256, 0, stream>>>(Wv,  wv16,  (long long)L_ * HH);
    k_cvt_f16<<<2048, 256, 0, stream>>>(Wo,  wo16,  (long long)L_ * HH);
    k_cvt_f16<<<2048, 256, 0, stream>>>(Wi,  wi16,  (long long)L_ * HFF);
    k_cvt_f16<<<2048, 256, 0, stream>>>(Wo2, wo216, (long long)L_ * HFF);
    k_cvt_f16<<<64,   256, 0, stream>>>(clfW, clfW16, (long long)H_ * NL_);

    k_maskbias<<<(B_ * S_ + 255) / 256, 256, 0, stream>>>(imask, mb, B_ * S_);
    k_embed<<<T_, 256, 0, stream>>>(ids, tids, wemb, pemb, temb, eg, ebt, x, x16);

    auto gemm = [&](bool exact,
                    const _Float16* A, long long lda, long long sAz,
                    const _Float16* Bm, long long ldb, long long sBz,
                    float* C, long long ldc, long long sCb, long long sCh,
                    _Float16* C16, long long ldc16, long long sC16b, long long sC16h,
                    int divC, int c16mode, const float* bias, int bias_mode,
                    int M, int N, int K, float alpha, int act, int Z) {
        GemmP p;
        p.A = A; p.lda = lda; p.sAz = sAz;
        p.Bm = Bm; p.ldb = ldb; p.sBz = sBz;
        p.C = C; p.ldc = ldc; p.sCb = sCb; p.sCh = sCh;
        p.C16 = C16; p.ldc16 = ldc16; p.sC16b = sC16b; p.sC16h = sC16h;
        p.divC = divC; p.c16mode = c16mode; p.bias = bias; p.bias_mode = bias_mode;
        p.M = M; p.N = N; p.K = K; p.alpha = alpha; p.act = act;
        dim3 grid((N + 63) / 64, (M + 127) / 128, Z);
        if (exact) k_gemm<true><<<grid, 256, 0, stream>>>(p);
        else       k_gemm<false><<<grid, 256, 0, stream>>>(p);
    };

    const float scale = 0.125f; // 1/sqrt(64)
    const int Zh = B_ * NH_;

    for (int l = 0; l < L_; ++l) {
        const _Float16* wq_l  = wq16  + (long long)l * HH;
        const _Float16* wk_l  = wk16  + (long long)l * HH;
        const _Float16* wv_l  = wv16  + (long long)l * HH;
        const _Float16* wo_l  = wo16  + (long long)l * HH;
        const _Float16* wi_l  = wi16  + (long long)l * HFF;
        const _Float16* wo2_l = wo216 + (long long)l * HFF;

        // Q/K/V projections written directly into f16 head layouts
        gemm(true, x16, H_, 0, wq_l, H_, 0, nullptr, 0, 0, 0,
             q16, 0, 0, 0, 1, 1, bq + (long long)l * H_, 1, T_, H_, H_, 1.0f, 0, 1);
        gemm(true, x16, H_, 0, wk_l, H_, 0, nullptr, 0, 0, 0,
             k16T, 0, 0, 0, 1, 2, bk + (long long)l * H_, 1, T_, H_, H_, 1.0f, 0, 1);
        gemm(true, x16, H_, 0, wv_l, H_, 0, nullptr, 0, 0, 0,
             v16b, 0, 0, 0, 1, 1, bv + (long long)l * H_, 1, T_, H_, H_, 1.0f, 0, 1);

        // scores = scale * Q @ K^T + mask_bias  (batched over B*NH)
        gemm(true, q16, D_, (long long)S_ * D_, k16T, S_, (long long)D_ * S_,
             scores, S_, (long long)S_ * S_, 0, nullptr, 0, 0, 0, 1, 0,
             mb, 2, S_, S_, D_, scale, 0, Zh);

        // softmax over keys -> f16 probs
        k_softmax<<<Zh * S_, 256, 0, stream>>>(scores, S_, probs, nullptr, S_, S_);

        // ctx = probs @ V, head-merged output [B,S,H] (f32 + f16)
        gemm(true, probs, S_, (long long)S_ * S_, v16b, D_, (long long)S_ * D_,
             ctx, H_, (long long)S_ * H_, D_,
             ctx16, H_, (long long)S_ * H_, D_,
             NH_, 0, nullptr, 0, S_, D_, S_, 1.0f, 0, Zh);

        // attention out-proj + residual LN
        gemm(true, ctx16, H_, 0, wo_l, H_, 0, tmp, H_, 0, 0, nullptr, 0, 0, 0, 1, 0,
             bo + (long long)l * H_, 1, T_, H_, H_, 1.0f, 0, 1);
        k_ln<<<T_, 256, 0, stream>>>(tmp, x, g1 + (long long)l * H_, b1 + (long long)l * H_, x, x16);

        // FFN: gelu(x @ Wi + bi) -> f16 directly
        gemm(true, x16, H_, 0, wi_l, FF_, 0, nullptr, 0, 0, 0,
             ff16, FF_, 0, 0, 1, 0,
             bi + (long long)l * FF_, 1, T_, FF_, H_, 1.0f, 1, 1);
        gemm(true, ff16, FF_, 0, wo2_l, H_, 0, tmp, H_, 0, 0, nullptr, 0, 0, 0, 1, 0,
             bo2 + (long long)l * H_, 1, T_, H_, FF_, 1.0f, 0, 1);
        k_ln<<<T_, 256, 0, stream>>>(tmp, x, g2 + (long long)l * H_, b2 + (long long)l * H_, x, x16);
    }

    // stable compaction of valid tokens, zero-padded
    k_compact<<<B_, 256, 0, stream>>>(vmask, x16, comp16);

    // classifier + final softmax (f32 output)
    gemm(false, comp16, H_, 0, clfW16, NL_, 0, logits, NL_, 0, 0, nullptr, 0, 0, 0, 1, 0,
         clfb, 1, T_, NL_, H_, 1.0f, 0, 1);
    k_softmax<<<T_, 256, 0, stream>>>(logits, NL_, nullptr, (float*)d_out, NL_, NL_);
}
